// RCNN_14697378087271
// MI455X (gfx1250) — compile-verified, matrix operand order checked by reading, not run on hardware
//
#include <hip/hip_runtime.h>
#include <stdint.h>

#define IMG_SZ   800
#define FH       200
#define FW       200
#define S_BIN    14
#define C_CH     256
#define NROI     512
#define NCLS     81
#define HID      1024
#define D1       (C_CH * S_BIN * S_BIN)   // 50176

typedef __attribute__((ext_vector_type(16))) __bf16 v16bf;
typedef __attribute__((ext_vector_type(8)))  float  v8f;

// one v_cvt_pk_bf16_f32: dst[15:0]=bf16(lo), dst[31:16]=bf16(hi)
static __device__ __forceinline__ uint32_t pkbf(float lo, float hi) {
    uint32_t r;
    asm("v_cvt_pk_bf16_f32 %0, %1, %2" : "=v"(r) : "v"(lo), "v"(hi));
    return r;
}

// -------------------------------------------------------------------------
// Kernel 1: ROI-align with bilinear sampling -> x [NROI][D1] in bf16
// grid (196, 512), block 256 : one output element per thread (196*256 = D1)
// -------------------------------------------------------------------------
__global__ __launch_bounds__(256) void roi_align_bf16(
    const float* __restrict__ feat,      // [C,200,200]
    const float* __restrict__ props,     // [NROI,4]
    __bf16* __restrict__ x)              // [NROI,D1]
{
    const int n = blockIdx.y;
    const int k = blockIdx.x * 256 + threadIdx.x;        // 0..D1-1
    const int c   = k / (S_BIN * S_BIN);
    const int pos = k - c * (S_BIN * S_BIN);
    const int iy  = pos / S_BIN;
    const int ix  = pos - iy * S_BIN;

    const float p0 = props[n * 4 + 0] * 0.25f;
    const float p1 = props[n * 4 + 1] * 0.25f;
    const float p2 = props[n * 4 + 2] * 0.25f;
    const float p3 = props[n * 4 + 3] * 0.25f;

    const float tx = (ix + 0.5f) / (float)S_BIN;
    const float ty = (iy + 0.5f) / (float)S_BIN;
    const float xs = p0 + (p2 - p0) * tx;
    const float ys = p1 + (p3 - p1) * ty;

    const float x0f = floorf(xs);
    const float y0f = floorf(ys);
    const float fx  = xs - x0f;
    const float fy  = ys - y0f;
    int x0 = (int)x0f; x0 = x0 < 0 ? 0 : (x0 > FW - 1 ? FW - 1 : x0);
    int x1 = x0 + 1;   x1 = x1 > FW - 1 ? FW - 1 : x1;
    int y0 = (int)y0f; y0 = y0 < 0 ? 0 : (y0 > FH - 1 ? FH - 1 : y0);
    int y1 = y0 + 1;   y1 = y1 > FH - 1 ? FH - 1 : y1;

    const float* f = feat + (size_t)c * (FH * FW);
    const float v00 = f[y0 * FW + x0];
    const float v01 = f[y0 * FW + x1];
    const float v10 = f[y1 * FW + x0];
    const float v11 = f[y1 * FW + x1];
    const float v = (1.f - fx) * (1.f - fy) * v00 + fx * (1.f - fy) * v01
                  + (1.f - fx) * fy * v10 + fx * fy * v11;
    x[(size_t)n * D1 + k] = (__bf16)v;
}

// ---- W staging helpers ---------------------------------------------------
static __device__ __forceinline__ void stage_w_aligned(
    const float* __restrict__ Wp, int N, __bf16 (*wt)[40], int nb, int kp)
{
    const float4 r0a = *(const float4*)(Wp);
    const float4 r0b = *(const float4*)(Wp + 4);
    const float4 r1a = *(const float4*)(Wp + N);
    const float4 r1b = *(const float4*)(Wp + N + 4);
    *(uint32_t*)(&wt[nb + 0][kp]) = pkbf(r0a.x, r1a.x);
    *(uint32_t*)(&wt[nb + 1][kp]) = pkbf(r0a.y, r1a.y);
    *(uint32_t*)(&wt[nb + 2][kp]) = pkbf(r0a.z, r1a.z);
    *(uint32_t*)(&wt[nb + 3][kp]) = pkbf(r0a.w, r1a.w);
    *(uint32_t*)(&wt[nb + 4][kp]) = pkbf(r0b.x, r1b.x);
    *(uint32_t*)(&wt[nb + 5][kp]) = pkbf(r0b.y, r1b.y);
    *(uint32_t*)(&wt[nb + 6][kp]) = pkbf(r0b.z, r1b.z);
    *(uint32_t*)(&wt[nb + 7][kp]) = pkbf(r0b.w, r1b.w);
}

static __device__ __forceinline__ void stage_w_guarded(
    const float* __restrict__ Wp, int N, int blockCol,
    __bf16 (*wt)[40], int wcBase, int wr)
{
#pragma unroll
    for (int j = 0; j < 16; ++j) {
        const int colb = blockCol + wcBase + j;
        const int cc   = colb < N ? colb : N - 1;   // clamp: branch-free
        float wv = Wp[cc];                          // always-valid load
        wv = colb < N ? wv : 0.f;                   // v_cndmask
        wt[wcBase + j][wr] = (__bf16)wv;
    }
}

// -------------------------------------------------------------------------
// Kernel 2: bf16 WMMA GEMM  out = act( A[MxK]bf16 * W[KxN]f32 + bias )
// Block: 256 threads (8 waves). Block tile 64(M) x 128(N), K step 32.
// Double-buffered LDS; A tile via GLOBAL_LOAD_ASYNC_TO_LDS_B128 (ASYNCcnt),
// W tile fp32 -> packed bf16 pairs (v_cvt_pk_bf16_f32) transposed in LDS.
// One barrier per K step; next-tile staging overlaps the 4 WMMAs.
// ALIGNED=true requires N % 128 == 0.
// -------------------------------------------------------------------------
template <bool ALIGNED, bool OUTBF, bool RELU>
__global__ __launch_bounds__(256) void gemm_bf16_wmma(
    const __bf16* __restrict__ A,      // [M x K] bf16 row-major
    const float*  __restrict__ W,      // [K x N] f32 row-major
    const float*  __restrict__ bias,   // [N]
    __bf16* __restrict__ outBf,        // bf16 output (OUTBF)
    float*  __restrict__ outF,         // f32 output (!OUTBF)
    int M, int N, int K)
{
    __shared__ __bf16 As[2][64][40];   // double buffered, +8 pad keeps 16B align
    __shared__ __bf16 Wt[2][128][40];  // transposed: [n][k]

    const int tid   = threadIdx.x;
    const int lane  = tid & 31;
    const int wid   = tid >> 5;
    const int waveM = wid & 1;         // 2 waves along M (32 rows each)
    const int waveN = wid >> 1;        // 4 waves along N (32 cols each)
    const int half  = lane >> 4;       // ISA bf16 fragment: K-half select
    const int lr    = lane & 15;
    const int kq    = half * 8;

    const int blockRow = blockIdx.y * 64;
    const int blockCol = blockIdx.x * 128;

    v8f acc[2][2];
#pragma unroll
    for (int i = 0; i < 2; ++i)
#pragma unroll
        for (int j = 0; j < 2; ++j)
#pragma unroll
            for (int g = 0; g < 8; ++g) acc[i][j][g] = 0.f;

    // A-tile mapping: 64 rows x 64B/row; 4 threads/row x 16B async copy
    const int ar = tid >> 2;
    const int ac = (tid & 3) * 8;
    const uint32_t ldsA0 = (uint32_t)(uintptr_t)(&As[0][ar][ac]); // low 32 = LDS off
    const uint32_t ldsA1 = (uint32_t)(uintptr_t)(&As[1][ar][ac]);
    uint64_t gaA = (uint64_t)(const void*)(A + (size_t)(blockRow + ar) * (size_t)K + ac);

    // W-tile mapping (ALIGNED): thread = 2 k-rows x 8 n-cols, packed stores
    const int kp = (tid >> 4) * 2;          // 0,2,..,30
    const int nb = (tid & 15) * 8;          // 0..120
    // W-tile mapping (guarded): thread = 1 k-row x 16 n-cols, clamped scalar
    const int wr     = tid >> 3;            // 0..31
    const int wcBase = (tid & 7) * 16;      // 0..112

    const float* WpAl = W + (size_t)kp * (size_t)N + blockCol + nb;  // ALIGNED
    const float* WpGd = W + (size_t)wr * (size_t)N;                  // guarded

    const int nk = K >> 5;

    // ---- prologue: stage buffer 0 (k = 0) ----
    asm volatile("global_load_async_to_lds_b128 %0, %1, off"
                 :: "v"(ldsA0), "v"(gaA) : "memory");
    if (ALIGNED) stage_w_aligned(WpAl, N, Wt[0], nb, kp);
    else         stage_w_guarded(WpGd, N, blockCol, Wt[0], wcBase, wr);

    for (int it = 0; it < nk; ++it) {
        const int b = it & 1;

        // current buffer's async A must be resident, and all waves must have
        // finished both staging (writes) and the previous compute (reads).
        asm volatile("s_wait_asynccnt 0x0" ::: "memory");
        __syncthreads();

        // ---- stage next buffer while we compute this one ----
        if (it + 1 < nk) {
            gaA += 64;                                  // 32 bf16 = 64 bytes
            const uint32_t dst = b ? ldsA0 : ldsA1;
            asm volatile("global_load_async_to_lds_b128 %0, %1, off"
                         :: "v"(dst), "v"(gaA) : "memory");
            if (ALIGNED) {
                WpAl += (size_t)32 * N;
                stage_w_aligned(WpAl, N, Wt[b ^ 1], nb, kp);
                if (it + 2 < nk)
                    __builtin_prefetch((const void*)(WpAl + (size_t)32 * N), 0, 1);
            } else {
                WpGd += (size_t)32 * N;
                stage_w_guarded(WpGd, N, blockCol, Wt[b ^ 1], wcBase, wr);
                if (it + 2 < nk)
                    __builtin_prefetch((const void*)(WpGd + (size_t)32 * N + blockCol + wcBase), 0, 1);
            }
        }

        // ---- build fragments per ISA 7.12.2 bf16 layout ----
        union Frag { uint4 u[2]; v16bf v; };
        Frag a[2], bfr[2];
#pragma unroll
        for (int t = 0; t < 2; ++t) {
            const int row = waveM * 32 + t * 16 + lr;
            a[t].u[0] = *(const uint4*)(&As[b][row][kq]);       // K = kq..kq+7
            a[t].u[1] = *(const uint4*)(&As[b][row][16 + kq]);  // K = 16+kq..
            const int col = waveN * 32 + t * 16 + lr;
            bfr[t].u[0] = *(const uint4*)(&Wt[b][col][kq]);
            bfr[t].u[1] = *(const uint4*)(&Wt[b][col][16 + kq]);
        }
#pragma unroll
        for (int i = 0; i < 2; ++i)
#pragma unroll
            for (int j = 0; j < 2; ++j)
                acc[i][j] = __builtin_amdgcn_wmma_f32_16x16x32_bf16(
                    false, a[i].v, false, bfr[j].v, (short)0, acc[i][j], false, false);
        // no trailing barrier: next iteration's barrier orders buffer reuse
    }

    // ---- epilogue: bias, relu, store (C/D layout: N=lane&15, M=half*8+g) ----
#pragma unroll
    for (int i = 0; i < 2; ++i) {
#pragma unroll
        for (int j = 0; j < 2; ++j) {
            const int col = blockCol + waveN * 32 + j * 16 + lr;
            if (ALIGNED || col < N) {
                const float bv = bias[col];
#pragma unroll
                for (int g = 0; g < 8; ++g) {
                    const int row = blockRow + waveM * 32 + i * 16 + half * 8 + g;
                    float v = acc[i][j][g] + bv;
                    if (RELU) v = v > 0.f ? v : 0.f;
                    if (OUTBF) outBf[(size_t)row * N + col] = (__bf16)v;
                    else       outF [(size_t)row * N + col] = v;
                }
            }
        }
    }
    (void)M;
}

// -------------------------------------------------------------------------
// Kernel 3: softmax(81) -> argmax fg -> score/label -> box decode
// -------------------------------------------------------------------------
__global__ __launch_bounds__(256) void head_kernel(
    const float* __restrict__ logits,   // [NROI,81]
    const float* __restrict__ deltas,   // [NROI,81*4]
    const float* __restrict__ props,    // [NROI,4]
    float* __restrict__ boxes,          // [NROI,4]
    float* __restrict__ scores,         // [NROI]
    int*   __restrict__ labels)         // [NROI]
{
    const int n = blockIdx.x * 256 + threadIdx.x;
    if (n >= NROI) return;

    const float* l = logits + (size_t)n * NCLS;
    float m = l[0];
    for (int c = 1; c < NCLS; ++c) m = fmaxf(m, l[c]);
    float s = 0.f;
    for (int c = 0; c < NCLS; ++c) s += __expf(l[c] - m);
    const float inv = 1.f / s;

    float best = -1.f; int bc = 1;
    for (int c = 1; c < NCLS; ++c) {
        const float p = __expf(l[c] - m) * inv;
        if (p > best) { best = p; bc = c; }
    }

    const float* d = deltas + (size_t)n * (NCLS * 4) + bc * 4;
    const float r0 = props[n * 4 + 0], r1 = props[n * 4 + 1];
    const float r2 = props[n * 4 + 2], r3 = props[n * 4 + 3];
    const float w = r2 - r0, h = r3 - r1;
    const float cx = r0 + 0.5f * w, cy = r1 + 0.5f * h;
    const float px = cx + d[0] * w, py = cy + d[1] * h;
    const float dw = fminf(fmaxf(d[2], -4.f), 4.f);
    const float dh = fminf(fmaxf(d[3], -4.f), 4.f);
    const float pw = w * __expf(dw), ph = h * __expf(dh);
    float b0 = px - 0.5f * pw, b1 = py - 0.5f * ph;
    float b2 = px + 0.5f * pw, b3 = py + 0.5f * ph;
    const float hi = (float)(IMG_SZ - 1);
    b0 = fminf(fmaxf(b0, 0.f), hi); b1 = fminf(fmaxf(b1, 0.f), hi);
    b2 = fminf(fmaxf(b2, 0.f), hi); b3 = fminf(fmaxf(b3, 0.f), hi);

    boxes[n * 4 + 0] = b0; boxes[n * 4 + 1] = b1;
    boxes[n * 4 + 2] = b2; boxes[n * 4 + 3] = b3;
    scores[n] = best; labels[n] = bc;
}

// -------------------------------------------------------------------------
// Kernel 4: stable descending rank-sort (N=512, O(N^2) is trivial)
// -------------------------------------------------------------------------
__global__ __launch_bounds__(256) void sort_kernel(
    const float* __restrict__ boxes, const float* __restrict__ scores,
    const int* __restrict__ labels,
    float* __restrict__ sb, float* __restrict__ ss, int* __restrict__ sl)
{
    const int n = blockIdx.x * 256 + threadIdx.x;
    if (n >= NROI) return;
    const float s = scores[n];
    int r = 0;
    for (int j = 0; j < NROI; ++j) {
        const float sj = scores[j];
        r += (sj > s) || (sj == s && j < n);   // stable: ties by index
    }
    sb[r * 4 + 0] = boxes[n * 4 + 0];
    sb[r * 4 + 1] = boxes[n * 4 + 1];
    sb[r * 4 + 2] = boxes[n * 4 + 2];
    sb[r * 4 + 3] = boxes[n * 4 + 3];
    ss[r] = s; sl[r] = labels[n];
}

// -------------------------------------------------------------------------
// Kernel 5: sequential NMS, one 512-thread block, boxes+keep in LDS
// -------------------------------------------------------------------------
__global__ __launch_bounds__(512) void nms_kernel(
    const float* __restrict__ sb, const float* __restrict__ ss,
    int* __restrict__ keep)
{
    __shared__ float bx[NROI][4];
    __shared__ int   kp[NROI];
    const int t = threadIdx.x;
    bx[t][0] = sb[t * 4 + 0]; bx[t][1] = sb[t * 4 + 1];
    bx[t][2] = sb[t * 4 + 2]; bx[t][3] = sb[t * 4 + 3];
    kp[t] = (ss[t] > 0.05f) ? 1 : 0;
    __syncthreads();

    const float a0 = bx[t][0], a1 = bx[t][1], a2 = bx[t][2], a3 = bx[t][3];
    const float areaT = fmaxf(a2 - a0, 0.f) * fmaxf(a3 - a1, 0.f);

    for (int i = 0; i < NROI; ++i) {
        if (kp[i] && t > i && kp[t]) {
            const float i0 = bx[i][0], i1 = bx[i][1], i2 = bx[i][2], i3 = bx[i][3];
            const float areaI = fmaxf(i2 - i0, 0.f) * fmaxf(i3 - i1, 0.f);
            const float xx0 = fmaxf(i0, a0), yy0 = fmaxf(i1, a1);
            const float xx1 = fminf(i2, a2), yy1 = fminf(i3, a3);
            const float inter = fmaxf(xx1 - xx0, 0.f) * fmaxf(yy1 - yy0, 0.f);
            const float iou = inter / (areaI + areaT - inter + 1e-6f);
            if (iou > 0.5f) kp[t] = 0;
        }
        __syncthreads();
    }
    keep[t] = kp[t];
}

// -------------------------------------------------------------------------
// Kernel 6: final = concat(boxes, score)*keep ; labels ; keep
// d_out: 512*5 f32, then 512 int32 labels, then 512 int32 keep
// -------------------------------------------------------------------------
__global__ __launch_bounds__(256) void final_kernel(
    const float* __restrict__ sb, const float* __restrict__ ss,
    const int* __restrict__ sl, const int* __restrict__ keep,
    float* __restrict__ out)
{
    const int n = blockIdx.x * 256 + threadIdx.x;
    if (n >= NROI) return;
    const float kf = keep[n] ? 1.f : 0.f;
    out[n * 5 + 0] = sb[n * 4 + 0] * kf;
    out[n * 5 + 1] = sb[n * 4 + 1] * kf;
    out[n * 5 + 2] = sb[n * 4 + 2] * kf;
    out[n * 5 + 3] = sb[n * 4 + 3] * kf;
    out[n * 5 + 4] = ss[n] * kf;
    int* oi = (int*)out;
    oi[NROI * 5 + n]        = sl[n];
    oi[NROI * 5 + NROI + n] = keep[n];
}

// -------------------------------------------------------------------------
extern "C" void kernel_launch(void* const* d_in, const int* in_sizes, int n_in,
                              void* d_out, int out_size, void* d_ws, size_t ws_size,
                              hipStream_t stream)
{
    (void)in_sizes; (void)n_in; (void)out_size; (void)ws_size;
    const float* features  = (const float*)d_in[0];
    const float* proposals = (const float*)d_in[1];
    const float* w1    = (const float*)d_in[2];
    const float* b1    = (const float*)d_in[3];
    const float* w2    = (const float*)d_in[4];
    const float* b2    = (const float*)d_in[5];
    const float* w_cls = (const float*)d_in[6];
    const float* b_cls = (const float*)d_in[7];
    const float* w_reg = (const float*)d_in[8];
    const float* b_reg = (const float*)d_in[9];

    char* ws = (char*)d_ws;
    size_t off = 0;
    auto alloc = [&](size_t bytes) -> void* {
        off = (off + 255) & ~(size_t)255;
        void* p = ws + off;
        off += bytes;
        return p;
    };

    __bf16* xb     = (__bf16*)alloc((size_t)NROI * D1 * 2);        // ~51.4 MB
    __bf16* h1     = (__bf16*)alloc((size_t)NROI * HID * 2);       // 1 MB
    __bf16* h2     = (__bf16*)alloc((size_t)NROI * HID * 2);       // 1 MB
    float* logits  = (float*)alloc((size_t)NROI * NCLS * 4);
    float* deltas  = (float*)alloc((size_t)NROI * NCLS * 4 * 4);
    float* boxes   = (float*)alloc((size_t)NROI * 4 * 4);
    float* scores  = (float*)alloc((size_t)NROI * 4);
    int*   labels  = (int*)  alloc((size_t)NROI * 4);
    float* sboxes  = (float*)alloc((size_t)NROI * 4 * 4);
    float* sscores = (float*)alloc((size_t)NROI * 4);
    int*   slabels = (int*)  alloc((size_t)NROI * 4);
    int*   keep    = (int*)  alloc((size_t)NROI * 4);

    // 1) ROI-align -> bf16 activations
    roi_align_bf16<<<dim3(D1 / 256, NROI), 256, 0, stream>>>(features, proposals, xb);

    // 2) FC1: [512,50176] x [50176,1024] + b1, ReLU -> bf16
    gemm_bf16_wmma<true, true, true><<<dim3(HID / 128, NROI / 64), 256, 0, stream>>>(
        xb, w1, b1, h1, nullptr, NROI, HID, D1);

    // 3) FC2: [512,1024] x [1024,1024] + b2, ReLU -> bf16
    gemm_bf16_wmma<true, true, true><<<dim3(HID / 128, NROI / 64), 256, 0, stream>>>(
        h1, w2, b2, h2, nullptr, NROI, HID, HID);

    // 4) cls head: [512,1024] x [1024,81] + b_cls -> f32 logits
    gemm_bf16_wmma<false, false, false><<<dim3(1, NROI / 64), 256, 0, stream>>>(
        h2, w_cls, b_cls, nullptr, logits, NROI, NCLS, HID);

    // 5) reg head: [512,1024] x [1024,324] + b_reg -> f32 deltas
    gemm_bf16_wmma<false, false, false><<<dim3(3, NROI / 64), 256, 0, stream>>>(
        h2, w_reg, b_reg, nullptr, deltas, NROI, NCLS * 4, HID);

    // 6) softmax / argmax / decode
    head_kernel<<<dim3(NROI / 256), 256, 0, stream>>>(
        logits, deltas, proposals, boxes, scores, labels);

    // 7) stable descending sort by score
    sort_kernel<<<dim3(NROI / 256), 256, 0, stream>>>(
        boxes, scores, labels, sboxes, sscores, slabels);

    // 8) NMS
    nms_kernel<<<1, NROI, 0, stream>>>(sboxes, sscores, keep);

    // 9) final masked outputs
    final_kernel<<<dim3(NROI / 256), 256, 0, stream>>>(
        sboxes, sscores, slabels, keep, (float*)d_out);
}
